// HeteroGraphGAT_29892972380356
// MI455X (gfx1250) — compile-verified, matrix operand order checked by reading, not run on hardware
//
#include <hip/hip_runtime.h>

// ---------------------------------------------------------------------------
// HeteroGraphGAT for MI455X (gfx1250, wave32, WMMA)
//   C=256 channels, H=8 heads, D=32, ED=16, L=2 layers
// ---------------------------------------------------------------------------

#define CCH 256
#define HH  8
#define DD  32
#define EDIM 16

typedef __attribute__((ext_vector_type(16))) __bf16   v16bf;
typedef __attribute__((ext_vector_type(8)))  float    v8f;
typedef __attribute__((ext_vector_type(4)))  unsigned v4u;

struct bf16x16 { v4u lo, hi; };

__device__ __forceinline__ unsigned short f2bf(float f) {
    unsigned u = __float_as_uint(f);
    unsigned r = 0x7FFFu + ((u >> 16) & 1u);   // round-to-nearest-even
    return (unsigned short)((u + r) >> 16);
}

// ---- fp32 -> bf16 convert (grid-stride) -----------------------------------
__global__ void k_f32_to_bf16(const float* __restrict__ x,
                              unsigned short* __restrict__ y, long n) {
    long i = (long)blockIdx.x * blockDim.x + threadIdx.x;
    long s = (long)gridDim.x * blockDim.x;
    for (; i < n; i += s) y[i] = f2bf(x[i]);
}

// ---- W[256,256] fp32 -> Wt[256,256] bf16 transposed (Wt[n][k]=W[k][n]) ----
__global__ void k_transpose_bf16(const float* __restrict__ W,
                                 unsigned short* __restrict__ Wt) {
    int tid = blockIdx.x * blockDim.x + threadIdx.x;   // 65536 threads
    int k = tid >> 8, n = tid & 255;
    Wt[n * CCH + k] = f2bf(W[k * CCH + n]);
}

// ---- WMMA GEMM: Y[M,256] = Xb[M,256] @ W,  bf16 in / f32 accumulate -------
// block = 128 threads (4 waves); block b -> rows [16b,16b+16); wave w -> cols
// [64w, 64w+64) as 4 WMMA 16x16 tiles.  M must be a multiple of 16 (N=100000).
__global__ void k_gemm_wmma(const unsigned short* __restrict__ Xb,
                            const unsigned short* __restrict__ Wt,
                            float* __restrict__ Y, int M) {
    const int wave = threadIdx.x >> 5;
    const int lane = threadIdx.x & 31;
    const int m0   = blockIdx.x * 16;
    if (m0 >= M) return;
    const int row   = m0 + (lane & 15);
    const int khalf = (lane >> 4) * 8;   // lanes 16..31 take K 8..15 / 24..31

    v8f acc[4] = {};
    const unsigned short* xrow = Xb + (size_t)row * CCH;

    for (int k0 = 0; k0 < CCH; k0 += 32) {
        bf16x16 ar;
        ar.lo = *(const v4u*)(xrow + k0 + khalf);
        ar.hi = *(const v4u*)(xrow + k0 + khalf + 16);
        v16bf a = __builtin_bit_cast(v16bf, ar);
#pragma unroll
        for (int t = 0; t < 4; ++t) {
            int col = wave * 64 + t * 16 + (lane & 15);
            const unsigned short* wrow = Wt + (size_t)col * CCH;
            bf16x16 br;
            br.lo = *(const v4u*)(wrow + k0 + khalf);
            br.hi = *(const v4u*)(wrow + k0 + khalf + 16);
            v16bf b = __builtin_bit_cast(v16bf, br);
            acc[t] = __builtin_amdgcn_wmma_f32_16x16x32_bf16(
                false, a, false, b, (short)0, acc[t], false, false);
        }
    }
    // C/D layout: VGPR r -> M = r + 8*(lane>=16), N = lane&15
    const int rbase = m0 + ((lane >> 4) << 3);
#pragma unroll
    for (int t = 0; t < 4; ++t) {
        int col = wave * 64 + t * 16 + (lane & 15);
#pragma unroll
        for (int r = 0; r < 8; ++r)
            Y[(size_t)(rbase + r) * CCH + col] = acc[t][r];
    }
}

// ---- fold attention vector into W:  c[k,h] = sum_d W[k, h*32+d]*a[h*32+d] -
__global__ void k_fold(const float* __restrict__ W, const float* __restrict__ a,
                       int K, float* __restrict__ c) {
    int tid = blockIdx.x * blockDim.x + threadIdx.x;
    if (tid >= K * HH) return;
    int k = tid >> 3, h = tid & 7;
    const float* wr = W + (size_t)k * CCH + h * DD;
    const float* ar = a + h * DD;
    float s = 0.f;
#pragma unroll
    for (int d = 0; d < DD; ++d) s += wr[d] * ar[d];
    c[tid] = s;
}

// ---- skinny GEMM: s[rows,8] = X[rows,K] @ c[K,8] --------------------------
__global__ void k_skinny(const float* __restrict__ X, const float* __restrict__ c,
                         int rows, int K, float* __restrict__ s) {
    int tid = blockIdx.x * blockDim.x + threadIdx.x;
    if (tid >= rows * HH) return;
    int r = tid >> 3, h = tid & 7;
    const float* xr = X + (size_t)r * K;
    float acc = 0.f;
    for (int k = 0; k < K; ++k) acc += xr[k] * c[k * HH + h];
    s[tid] = acc;
}

// ---- edge pass A: alpha + leaky-relu + segment-max (encoded uint atomic) --
__global__ void k_edge_alpha(const int* __restrict__ ei,
                             const float* __restrict__ s_src,
                             const float* __restrict__ s_dst,
                             const float* __restrict__ s_e,
                             float* __restrict__ alpha,
                             unsigned* __restrict__ amax, int E) {
    int tid = blockIdx.x * blockDim.x + threadIdx.x;
    if (tid >= E * HH) return;
    int e = tid >> 3, h = tid & 7;
    int src = ei[e], dst = ei[E + e];
    float a = s_src[src * HH + h] + s_dst[dst * HH + h] + s_e[tid];
    a = a > 0.f ? a : 0.2f * a;                       // leaky_relu(0.2)
    alpha[tid] = a;
    unsigned u = __float_as_uint(a);
    unsigned enc = (u & 0x80000000u) ? ~u : (u | 0x80000000u); // monotone key
    atomicMax(&amax[dst * HH + h], enc);
}

// ---- edge pass B: ex = exp(alpha-amax); denom += ex; out += ex*hs[src] ----
// one wave per edge; lanes 0..7 own heads; all 32 lanes stream 256 channels.
__global__ void k_edge_scatter(const int* __restrict__ ei,
                               const float* __restrict__ alpha,
                               const unsigned* __restrict__ amax,
                               const float* __restrict__ hs,
                               float* __restrict__ denom,
                               float* __restrict__ out, int E) {
    int gw   = (blockIdx.x * blockDim.x + threadIdx.x) >> 5;
    int lane = threadIdx.x & 31;
    int nw   = (gridDim.x * blockDim.x) >> 5;
    for (int e = gw; e < E; e += nw) {
        int src = ei[e], dst = ei[E + e];
        float ex = 0.f;
        if (lane < HH) {
            unsigned enc = amax[dst * HH + lane];
            float mx = (enc & 0x80000000u) ? __uint_as_float(enc ^ 0x80000000u)
                                           : __uint_as_float(~enc);
            ex = __expf(alpha[e * HH + lane] - mx);
            atomicAdd(&denom[dst * HH + lane], ex);
        }
        const float* hrow = hs  + (size_t)src * CCH;
        float*       orow = out + (size_t)dst * CCH;
#pragma unroll
        for (int k = 0; k < HH; ++k) {
            float exk = __shfl(ex, k, 32);
            int c = k * DD + lane;                     // coalesced 128B
            atomicAdd(&orow[c], exk * hrow[c]);
        }
    }
}

// ---- node epilogue: /denom + bias, LayerNorm, ReLU (in-place ok) ----------
__global__ void k_epilogue(const float* __restrict__ raw,
                           const float* __restrict__ denom,
                           const float* __restrict__ b,
                           const float* __restrict__ lnw,
                           const float* __restrict__ lnb,
                           float* __restrict__ xout) {
    int n = blockIdx.x, c = threadIdx.x;               // 256 threads/node
    float v = raw[(size_t)n * CCH + c] /
              (denom[n * HH + (c >> 5)] + 1e-16f) + b[c];
    __shared__ float red[CCH];
    red[c] = v; __syncthreads();
    for (int s = 128; s > 0; s >>= 1) { if (c < s) red[c] += red[c + s]; __syncthreads(); }
    float mu = red[0] * (1.f / CCH);
    __syncthreads();
    float dv = v - mu;
    red[c] = dv * dv; __syncthreads();
    for (int s = 128; s > 0; s >>= 1) { if (c < s) red[c] += red[c + s]; __syncthreads(); }
    float y = dv * rsqrtf(red[0] * (1.f / CCH) + 1e-5f) * lnw[c] + lnb[c];
    xout[(size_t)n * CCH + c] = y > 0.f ? y : 0.f;
}

// ---------------------------------------------------------------------------
struct WS {
    float *xu1, *xi1, *hs, *alpha, *s_e, *s_src, *s_dst, *denom, *cs, *cd, *ce;
    unsigned *amax;
    unsigned short *xbf, *Wtbf;
};

static void run_pass(const float* x_src, const float* x_dst, const int* ei,
                     const float* ea, const float* Wsrc, const float* Wdst,
                     const float* We, const float* asrc, const float* adst,
                     const float* ae, const float* b, const float* lnw,
                     const float* lnb, float* x_out, const WS& w,
                     int N, int E, hipStream_t st) {
    // bf16 conversions + WMMA GEMM for hs
    k_transpose_bf16<<<(CCH * CCH) / 256, 256, 0, st>>>(Wsrc, w.Wtbf);
    k_f32_to_bf16<<<4096, 256, 0, st>>>(x_src, w.xbf, (long)N * CCH);
    k_gemm_wmma<<<N / 16, 128, 0, st>>>(w.xbf, w.Wtbf, w.hs, N);
    // folded attention projections
    k_fold<<<(CCH * HH + 127) / 128, 128, 0, st>>>(Wsrc, asrc, CCH, w.cs);
    k_fold<<<(CCH * HH + 127) / 128, 128, 0, st>>>(Wdst, adst, CCH, w.cd);
    k_fold<<<(EDIM * HH + 127) / 128, 128, 0, st>>>(We, ae, EDIM, w.ce);
    k_skinny<<<((long)N * HH + 255) / 256, 256, 0, st>>>(x_src, w.cs, N, CCH, w.s_src);
    k_skinny<<<((long)N * HH + 255) / 256, 256, 0, st>>>(x_dst, w.cd, N, CCH, w.s_dst);
    k_skinny<<<((long)E * HH + 255) / 256, 256, 0, st>>>(ea, w.ce, E, EDIM, w.s_e);
    // clear accumulators (0 is identity for encoded max / sum)
    hipMemsetAsync(w.amax,  0, (size_t)N * HH * 4, st);
    hipMemsetAsync(w.denom, 0, (size_t)N * HH * 4, st);
    hipMemsetAsync(x_out,   0, (size_t)N * CCH * 4, st);
    // edge passes
    k_edge_alpha<<<((long)E * HH + 255) / 256, 256, 0, st>>>(
        ei, w.s_src, w.s_dst, w.s_e, w.alpha, w.amax, E);
    k_edge_scatter<<<2048, 256, 0, st>>>(ei, w.alpha, w.amax, w.hs,
                                         w.denom, x_out, E);
    // normalize + bias + LN + ReLU (in place on x_out)
    k_epilogue<<<N, CCH, 0, st>>>(x_out, w.denom, b, lnw, lnb, x_out);
}

extern "C" void kernel_launch(void* const* d_in, const int* in_sizes, int n_in,
                              void* d_out, int out_size, void* d_ws, size_t ws_size,
                              hipStream_t stream) {
    const int N = in_sizes[0] / CCH;
    const int E = in_sizes[2] / 2;

    const float* x_user = (const float*)d_in[0];
    const float* x_item = (const float*)d_in[1];
    const int*   ei0    = (const int*)d_in[2];
    const int*   ei1    = (const int*)d_in[3];
    const float* ea0    = (const float*)d_in[4];
    const float* ea1    = (const float*)d_in[5];
    const float* P[14];
    for (int i = 0; i < 14; ++i) P[i] = (const float*)d_in[6 + i];
    const float* lnwu = (const float*)d_in[20];
    const float* lnbu = (const float*)d_in[21];
    const float* lnwi = (const float*)d_in[22];
    const float* lnbi = (const float*)d_in[23];

    // ---- carve workspace ----
    char* p = (char*)d_ws;
    auto alloc = [&](size_t bytes) -> void* {
        void* r = (void*)p; p += (bytes + 255) & ~(size_t)255; return r;
    };
    WS w;
    w.xu1   = (float*)alloc((size_t)N * CCH * 4);
    w.xi1   = (float*)alloc((size_t)N * CCH * 4);
    w.hs    = (float*)alloc((size_t)N * CCH * 4);
    w.xbf   = (unsigned short*)alloc((size_t)N * CCH * 2);
    w.Wtbf  = (unsigned short*)alloc((size_t)CCH * CCH * 2);
    w.alpha = (float*)alloc((size_t)E * HH * 4);
    w.s_e   = (float*)alloc((size_t)E * HH * 4);
    w.s_src = (float*)alloc((size_t)N * HH * 4);
    w.s_dst = (float*)alloc((size_t)N * HH * 4);
    w.denom = (float*)alloc((size_t)N * HH * 4);
    w.amax  = (unsigned*)alloc((size_t)N * HH * 4);
    w.cs    = (float*)alloc((size_t)CCH * HH * 4);
    w.cd    = (float*)alloc((size_t)CCH * HH * 4);
    w.ce    = (float*)alloc((size_t)EDIM * HH * 4);

    float* out_u = (float*)d_out;
    float* out_i = (float*)d_out + (size_t)N * CCH;

    for (int l = 0; l < 2; ++l) {
        const float* xu = (l == 0) ? x_user : w.xu1;
        const float* xi = (l == 0) ? x_item : w.xi1;
        float* nu = (l == 0) ? w.xu1 : out_u;
        float* ni = (l == 0) ? w.xi1 : out_i;
        const size_t oW = (size_t)l * CCH * CCH;   // [L,256,256]
        const size_t oE = (size_t)l * EDIM * CCH;  // [L,16,256]
        const size_t oV = (size_t)l * CCH;         // [L,8,32] and [L,256]

        // edge type 0: user -> item   (params P[0..6] = Wsrc0..b0)
        run_pass(xu, xi, ei0, ea0,
                 P[0] + oW, P[1] + oW, P[2] + oE,
                 P[3] + oV, P[4] + oV, P[5] + oV, P[6] + oV,
                 lnwi + oV, lnbi + oV, ni, w, N, E, stream);
        // edge type 1: item -> user   (params P[7..13] = Wsrc1..b1)
        run_pass(xi, xu, ei1, ea1,
                 P[7] + oW, P[8] + oW, P[9] + oE,
                 P[10] + oV, P[11] + oV, P[12] + oV, P[13] + oV,
                 lnwu + oV, lnbu + oV, nu, w, N, E, stream);
    }
}
